// NeuralDDEWithTime_46823733461186
// MI455X (gfx1250) — matrix operand order repaired and checked by reading, not compiled, not run on hardware
//
#include <hip/hip_runtime.h>
#include <hip/hip_bf16.h>

// ---------------------------------------------------------------------------
// Neural DDE (Dopri5, method of steps) for MI455X / gfx1250.
// 32 workgroups x 1 wave32; each wave owns 16 batch rows for the whole
// 600-step integration. All GEMMs via v_wmma_f32_16x16x32_f16 with weights
// resident in LDS. Delayed-history fetch uses GLOBAL_LOAD_ASYNC_TO_LDS_B128
// (ASYNCcnt) issued at step start and waited only before stage 1, hiding the
// L2 round trip behind stage-0 compute.
// ---------------------------------------------------------------------------

typedef __attribute__((ext_vector_type(16))) _Float16 v16h;
typedef __attribute__((ext_vector_type(8)))  _Float16 v8h;
typedef __attribute__((ext_vector_type(2)))  _Float16 v2h;
typedef __attribute__((ext_vector_type(8)))  float    v8f;
typedef __attribute__((ext_vector_type(2)))  float    v2f;

#define B_TOT   512
#define D_DIM   64
#define WID_N   128
#define S_STEPS 120
#define N_SEG   5
#define ROWS    16                                   // batch rows per wave (WMMA M)
#define KSTR    136                                  // padded f16 row stride (bank-conflict pad)
#define OSTR    ((N_SEG * S_STEPS + 1) * D_DIM)      // 601*64 floats per batch row

// Dopri5 tableau
__device__ const float ATAB[6][5] = {
    {0.f, 0.f, 0.f, 0.f, 0.f},
    {0.2f, 0.f, 0.f, 0.f, 0.f},
    {3.f/40.f, 9.f/40.f, 0.f, 0.f, 0.f},
    {44.f/45.f, -56.f/15.f, 32.f/9.f, 0.f, 0.f},
    {19372.f/6561.f, -25360.f/2187.f, 64448.f/6561.f, -212.f/729.f, 0.f},
    {9017.f/3168.f, -355.f/33.f, 46732.f/5247.f, 49.f/176.f, -5103.f/18656.f},
};
__device__ const float CTAB[6] = {0.f, 0.2f, 0.3f, 0.8f, 8.f/9.f, 1.f};
__device__ const float BTAB[6] = {35.f/384.f, 0.f, 500.f/1113.f, 125.f/192.f,
                                  -2187.f/6784.f, 11.f/84.f};

__device__ __forceinline__ void wait_lds() {
    asm volatile("s_wait_dscnt 0" ::: "memory");
}
__device__ __forceinline__ void wait_async() {
    asm volatile("s_wait_asynccnt 0" ::: "memory");
}

// A fragment: 16x32 f16. lane m (=lane&15), hi (=lane>>4): v0..3 hold
// K = hi*8 + 0..7, v4..7 hold K = 16 + hi*8 + 0..7 -> two ds_load_b128.
__device__ __forceinline__ v16h load_frag_A(const _Float16* buf, int m, int hi, int kc) {
    const int off = m * KSTR + kc * 32 + hi * 8;
    v8h lo = *(const v8h*)(buf + off);
    v8h hh = *(const v8h*)(buf + off + 16);
    v16h r;
#pragma unroll
    for (int i = 0; i < 8; ++i) { r[i] = lo[i]; r[i + 8] = hh[i]; }
    return r;
}

// B fragment: 32x16 f16 for column N = lane&15; v0..3 hold K = hi*8 + 0..7,
// v4..7 hold K = 16 + hi*8 + 0..7 within the kc chunk. Weights stored [N][K] row-major.
__device__ __forceinline__ v16h load_frag_B(const _Float16* w, int col, int hi, int nt, int kc) {
    const int off = (nt * 16 + col) * KSTR + kc * 32 + hi * 8;
    v8h lo = *(const v8h*)(w + off);
    v8h hh = *(const v8h*)(w + off + 16);
    v16h r;
#pragma unroll
    for (int i = 0; i < 8; ++i) { r[i] = lo[i]; r[i + 8] = hh[i]; }
    return r;
}

// One dense layer: D = act(A(16x128) * W^T + bias [+ t*tvec]).
// Output either relu->f16 into out_h (A layout for next layer) or raw f32 into out_f.
__device__ __forceinline__ void layer(const _Float16* abuf, const _Float16* wbuf,
                                      const float* bias, const float* tvec, float tval,
                                      int n_tiles,
                                      _Float16* out_h, float* out_f, bool do_relu,
                                      int lane) {
    const int m = lane & 15, hi = lane >> 4, col = m;
    for (int nt = 0; nt < n_tiles; ++nt) {
        float cinit = bias[nt * 16 + col];
        if (tvec) cinit += tval * tvec[nt * 16 + col];
        v8f acc;
#pragma unroll
        for (int i = 0; i < 8; ++i) acc[i] = cinit;   // bias per column => same in all 8 row-VGPRs
#pragma unroll
        for (int kc = 0; kc < 4; ++kc) {
            v16h a = load_frag_A(abuf, m, hi, kc);
            v16h b = load_frag_B(wbuf, col, hi, nt, kc);
            acc = __builtin_amdgcn_wmma_f32_16x16x32_f16(false, a, false, b,
                                                         (short)0, acc, false, false);
        }
#pragma unroll
        for (int v = 0; v < 8; ++v) {                  // C layout: M = v + 8*hi, N = nt*16+col
            float x = acc[v];
            if (do_relu) x = x > 0.f ? x : 0.f;
            const int M = v + 8 * hi;
            if (out_h) out_h[M * KSTR + nt * 16 + col] = (_Float16)x;
            else       out_f[M * D_DIM + nt * 16 + col] = x;
        }
    }
}

__global__ void __launch_bounds__(32, 1)
dde_dopri5_kernel(const float* __restrict__ ts, const float* __restrict__ y0,
                  const float* __restrict__ W1, const float* __restrict__ b1,
                  const float* __restrict__ W2, const float* __restrict__ b2,
                  const float* __restrict__ W3, const float* __restrict__ b3,
                  float* __restrict__ out) {
    __shared__ __align__(16) _Float16 xb [ROWS * KSTR];      // [y_eval | hist] / layer3 input
    __shared__ __align__(16) _Float16 h1b[ROWS * KSTR];      // layer1 out / layer2 in
    __shared__ __align__(16) _Float16 w1 [WID_N * KSTR];     // W1[:, :128]  (y|hist columns)
    __shared__ __align__(16) _Float16 w2 [WID_N * KSTR];
    __shared__ __align__(16) _Float16 w3 [(WID_N / 2) * KSTR];
    __shared__ float w1t[WID_N], b1l[WID_N], b2l[WID_N], b3l[D_DIM];
    __shared__ float ylds[ROWS * D_DIM];                     // current state y
    __shared__ float kall[6][ROWS * D_DIM];                  // RK stage derivatives
    __shared__ __align__(16) float hist[2][ROWS * D_DIM];    // delayed-grid double buffer

    const int lane = threadIdx.x;
    const int b0 = blockIdx.x * ROWS;

    const float ts0 = ts[0];
    const float dt  = ts[1] - ts[0];

    // ---- one-time weight staging (f32 -> f16, B-fragment-friendly [N][K] layout) ----
    for (int i = lane; i < WID_N * 128; i += 32) {
        const int n = i >> 7, k = i & 127;
        w1[n * KSTR + k] = (_Float16)W1[n * 129 + k];
        w2[n * KSTR + k] = (_Float16)W2[n * 128 + k];
    }
    for (int i = lane; i < (WID_N / 2) * 128; i += 32) {
        const int n = i >> 7, k = i & 127;
        w3[n * KSTR + k] = (_Float16)W3[n * 128 + k];
    }
    for (int n = lane; n < WID_N; n += 32) {
        w1t[n] = W1[n * 129 + 128];      // time-feature column (absorbed into C init)
        b1l[n] = b1[n];
        b2l[n] = b2[n];
    }
    for (int n = lane; n < D_DIM; n += 32) b3l[n] = b3[n];

    // ---- initial state: y = y0, hist[0] = y0 (segment-0 constant history), out[:,0,:] = y0
    for (int i = lane; i < ROWS * D_DIM; i += 32) {
        const int m = i >> 6, d = i & 63;
        const float v = y0[(size_t)(b0 + m) * D_DIM + d];
        ylds[i] = v;
        hist[0][i] = v;
        out[(size_t)(b0 + m) * OSTR + d] = v;
    }
    wait_lds();

    int cur = 0;
    for (int seg = 0; seg < N_SEG; ++seg) {
        for (int s = 0; s < S_STEPS; ++s) {
            const int g = seg * S_STEPS + s;
            const float t = ts0 + (float)g * dt;

            const float* hp0 = hist[cur];                 // yp0 (seg 0: constant y0)
            const float* hp1 = hp0;                       // yp1
            if (seg > 0) {
                // Async DMA yp1 (our own stored grid point s+1 of prev segment,
                // L2-hot) directly into LDS; consumed first at stage 1.
                float* hn = hist[cur ^ 1];
                const unsigned ldsb = (unsigned)(uintptr_t)(const void*)hn;
                const size_t base = (size_t)((seg - 1) * S_STEPS + s + 1) * D_DIM;
#pragma unroll
                for (int it = 0; it < 8; ++it) {          // 16B per lane per issue
                    const int cj = lane + it * 32;        // 256 x b128 chunks total
                    const int m = cj >> 4, q = cj & 15;
                    const float* ga = out + (size_t)(b0 + m) * OSTR + base + q * 4;
                    const unsigned la = ldsb + (unsigned)(m * 256 + q * 16);
                    asm volatile("global_load_async_to_lds_b128 %0, %1, off"
                                 :: "v"(la), "v"(ga) : "memory");
                }
                hp1 = hn;
            }
            wait_lds();

            // ---- six Dopri5 stages ----
            for (int st = 0; st < 6; ++st) {
                if (st == 1) wait_async();                // yp1 DMA must have landed
                const float c = CTAB[st];
                const float* hq = (st == 0) ? hp0 : hp1;  // stage 0 reads only yp0 (c==0)
                // build A operand: [ y + dt*sum_j a_sj * k_j  |  yp0 + c*(yp1-yp0) ]
                for (int i = lane; i < ROWS * D_DIM / 2; i += 32) {
                    const int m = i >> 5, d2 = (i & 31) * 2;
                    const int yo = m * D_DIM + d2;
                    float e0 = ylds[yo], e1 = ylds[yo + 1];
                    for (int j = 0; j < st; ++j) {
                        const float a = dt * ATAB[st][j];
                        e0 += a * kall[j][yo];
                        e1 += a * kall[j][yo + 1];
                    }
                    const float p0 = hp0[yo], p1 = hp0[yo + 1];
                    const float q0 = hq[yo],  q1 = hq[yo + 1];
                    const float h0 = p0 + c * (q0 - p0);
                    const float h1v = p1 + c * (q1 - p1);
                    v2h ye; ye[0] = (_Float16)e0; ye[1] = (_Float16)e1;
                    v2h hh; hh[0] = (_Float16)h0; hh[1] = (_Float16)h1v;
                    *(v2h*)(xb + m * KSTR + d2) = ye;
                    *(v2h*)(xb + m * KSTR + 64 + d2) = hh;
                }
                wait_lds();

                const float tv = t + c * dt;              // stage time feeds C init via w1t
                layer(xb,  w1, b1l, w1t,    tv,  8, h1b,     nullptr,  true,  lane);
                wait_lds();
                layer(h1b, w2, b2l, nullptr, 0.f, 8, xb,     nullptr,  true,  lane);
                wait_lds();
                layer(xb,  w3, b3l, nullptr, 0.f, 4, nullptr, kall[st], false, lane);
                wait_lds();
            }

            // ---- y_{n+1} = y + dt * (b . k) ; store state + required output ----
            for (int i = lane; i < ROWS * D_DIM / 2; i += 32) {
                const int m = i >> 5, d2 = (i & 31) * 2;
                const int yo = m * D_DIM + d2;
                float e0 = ylds[yo], e1 = ylds[yo + 1];
#pragma unroll
                for (int j = 0; j < 6; ++j) {
                    if (j == 1) continue;                 // b2 == 0
                    const float a = dt * BTAB[j];
                    e0 += a * kall[j][yo];
                    e1 += a * kall[j][yo + 1];
                }
                ylds[yo] = e0; ylds[yo + 1] = e1;
                v2f vv; vv[0] = e0; vv[1] = e1;
                *(v2f*)(out + (size_t)(b0 + m) * OSTR + (size_t)(g + 1) * D_DIM + d2) = vv;
            }
            wait_lds();
            if (seg > 0) cur ^= 1;                        // yp1 becomes next step's yp0
        }
    }

    // second tuple element: num_steps (int32) appended flat after ys
    if (blockIdx.x == 0 && lane == 0) {
        ((int*)out)[(size_t)B_TOT * OSTR] = N_SEG * S_STEPS;
    }
}

extern "C" void kernel_launch(void* const* d_in, const int* in_sizes, int n_in,
                              void* d_out, int out_size, void* d_ws, size_t ws_size,
                              hipStream_t stream) {
    (void)in_sizes; (void)n_in; (void)out_size; (void)d_ws; (void)ws_size;
    const float* ts = (const float*)d_in[0];
    const float* y0 = (const float*)d_in[1];
    const float* W1 = (const float*)d_in[2];
    const float* b1 = (const float*)d_in[3];
    const float* W2 = (const float*)d_in[4];
    const float* b2 = (const float*)d_in[5];
    const float* W3 = (const float*)d_in[6];
    const float* b3 = (const float*)d_in[7];
    float* out = (float*)d_out;

    dde_dopri5_kernel<<<dim3(B_TOT / ROWS), dim3(32), 0, stream>>>(
        ts, y0, W1, b1, W2, b2, W3, b3, out);
}
